// SelfAttention_11562051961377
// MI455X (gfx1250) — matrix operand order repaired and checked by reading, not compiled, not run on hardware
//
#include <hip/hip_runtime.h>
#include <hip/hip_bf16.h>
#include <math.h>

// Problem constants (from reference): B=4, C=256, H=W=64 -> N=4096, CD=C/8=32
#define BB 4
#define CC 256
#define NN 4096
#define CDIM 32

typedef __attribute__((ext_vector_type(16))) __bf16 v16bf;
typedef __attribute__((ext_vector_type(8)))  __bf16 v8bf;
typedef __attribute__((ext_vector_type(8)))  float  v8f;
typedef __attribute__((ext_vector_type(4)))  float  v4f;

// round-to-nearest-even f32 -> bf16 bit pattern
static __device__ __forceinline__ unsigned short f2bf(float f) {
  unsigned int u = __float_as_uint(f);
  unsigned int r = (u + 0x7FFFu + ((u >> 16) & 1u)) >> 16;
  return (unsigned short)r;
}

static __device__ __forceinline__ v16bf mk16(v8bf a, v8bf b) {
  return __builtin_shufflevector(a, b, 0,1,2,3,4,5,6,7,8,9,10,11,12,13,14,15);
}

static __device__ __forceinline__ v8f wmma_bf16(v16bf a, v16bf b, v8f c) {
  // D = A(16x32 bf16) * B(32x16 bf16) + C(16x16 f32)
  return __builtin_amdgcn_wmma_f32_16x16x32_bf16(false, a, false, b, (short)0, c,
                                                 false, false);
}

// ---------------------------------------------------------------------------
// Kernel 0: convert weights f32 -> bf16 into workspace
// ---------------------------------------------------------------------------
__global__ void __launch_bounds__(256) cvt_weights_kernel(
    const float* __restrict__ Wq, const float* __restrict__ Wk,
    const float* __restrict__ Wv,
    unsigned short* __restrict__ Wqb, unsigned short* __restrict__ Wkb,
    unsigned short* __restrict__ Wvb) {
  int i = blockIdx.x * blockDim.x + threadIdx.x;
  if (i < CDIM * CC) {
    Wqb[i] = f2bf(Wq[i]);
    Wkb[i] = f2bf(Wk[i]);
  }
  if (i < CC * CC) Wvb[i] = f2bf(Wv[i]);
}

// ---------------------------------------------------------------------------
// Kernel 1: projections via WMMA.
//   Qb [B][N][32]  bf16 : q[b,n,o]
//   Ktb[B][N][32]  bf16 : k[b,o,m] stored transposed as [m][o]
//   Vb [B][C][N]   bf16 : v[b,c,m]
// One block = 4 waves handles one (batch, 32-column) slab of x.
// ---------------------------------------------------------------------------
__global__ void __launch_bounds__(128) proj_kernel(
    const float* __restrict__ x,
    const unsigned short* __restrict__ Wqb,
    const unsigned short* __restrict__ Wkb,
    const unsigned short* __restrict__ Wvb,
    unsigned short* __restrict__ Qb,
    unsigned short* __restrict__ Ktb,
    unsigned short* __restrict__ Vb) {
  __shared__ __align__(16) unsigned short XT[32 * 256];  // XT[n][c] bf16, 16KB

  const int b  = blockIdx.x / (NN / 32);
  const int n0 = (blockIdx.x % (NN / 32)) * 32;
  const int t  = threadIdx.x;

  // Load + transpose x[b, :, n0:n0+32] into XT (bf16). Coalesced along n.
  {
    const int j  = t & 31;   // n within slab
    const int cb = t >> 5;   // 0..3
    const float* xb = x + (size_t)b * CC * NN + n0 + j;
    #pragma unroll 4
    for (int k = 0; k < 64; ++k) {
      int c = cb + (k << 2);
      XT[j * 256 + c] = f2bf(xb[(size_t)c * NN]);
    }
  }
  __syncthreads();

  const int wave = t >> 5;
  const int lane = t & 31;
  const int lm = lane & 15;   // col/row-within-tile
  const int lh = lane >> 4;   // half select

  // 40 output tiles: 0-3 Q, 4-7 K, 8-39 V.  10 tiles per wave.
  for (int i = 0; i < 10; ++i) {
    const int tile = wave * 10 + i;
    v8f acc = {};
    if (tile < 8) {
      // Q or K projection: out[n,o] = sum_c XT[n,c] * W[o,c]
      const bool isQ = tile < 4;
      const int tt = tile & 3;
      const int ntile = tt >> 1, ot = tt & 1;
      const unsigned short* W = isQ ? Wqb : Wkb;
      const unsigned short* arow = &XT[(ntile * 16 + lm) * 256];          // A: XT row n
      const unsigned short* brow = &W[(ot * 16 + lm) * 256 + lh * 16];    // B: W row o
      #pragma unroll
      for (int k0 = 0; k0 < 256; k0 += 32) {
        v16bf A = mk16(*(const v8bf*)(arow + k0 + lh * 16),
                       *(const v8bf*)(arow + k0 + lh * 16 + 8));
        v16bf Bm = mk16(*(const v8bf*)(brow + k0),
                        *(const v8bf*)(brow + k0 + 8));
        acc = wmma_bf16(A, Bm, acc);
      }
      unsigned short* dst = (isQ ? Qb : Ktb) +
          ((size_t)b * NN + n0 + ntile * 16) * CDIM + ot * 16 + lm;
      #pragma unroll
      for (int r = 0; r < 8; ++r) dst[(size_t)(r + 8 * lh) * CDIM] = f2bf(acc[r]);
    } else {
      // V projection: out[c,n] = sum_c' Wv[c,c'] * XT[n,c']
      const int idx = tile - 8;
      const int ct = idx >> 1, ntile = idx & 1;
      const unsigned short* arow = &Wvb[(ct * 16 + lm) * 256];             // A: Wv row c
      const unsigned short* brow = &XT[(ntile * 16 + lm) * 256 + lh * 16]; // B: XT row n
      #pragma unroll
      for (int k0 = 0; k0 < 256; k0 += 32) {
        v16bf A = mk16(*(const v8bf*)(arow + k0 + lh * 16),
                       *(const v8bf*)(arow + k0 + lh * 16 + 8));
        v16bf Bm = mk16(*(const v8bf*)(brow + k0),
                        *(const v8bf*)(brow + k0 + 8));
        acc = wmma_bf16(A, Bm, acc);
      }
      unsigned short* dst =
          Vb + ((size_t)b * CC + ct * 16) * NN + n0 + ntile * 16 + lm;
      #pragma unroll
      for (int r = 0; r < 8; ++r) dst[(size_t)(r + 8 * lh) * NN] = f2bf(acc[r]);
    }
  }
}

// ---------------------------------------------------------------------------
// Kernel 2: fused flash attention + epilogue y = gamma*O^T + x.
// One block = 4 waves = (batch b, 16-query tile). Wave 0 computes S + online
// softmax ONCE per 32-key step and publishes P/rescale factors through
// double-buffered LDS (single barrier per step); every wave then runs 4 PV
// WMMAs on its private 64-column slice of V.
// ---------------------------------------------------------------------------
__global__ void __launch_bounds__(128) attn_kernel(
    const float* __restrict__ x, const float* __restrict__ gamma,
    const unsigned short* __restrict__ Qb,
    const unsigned short* __restrict__ Ktb,
    const unsigned short* __restrict__ Vb,
    float* __restrict__ y) {
  __shared__ __align__(16) float          Sls[16 * 32];       // wave0-private scores
  __shared__ __align__(16) unsigned short Pls[2][16 * 32];    // probs bf16, dbl-buf
  __shared__ __align__(16) float          fac[2][16];         // row rescale factors
  __shared__ __align__(16) float          Ols[256 * 16];      // O transpose staging

  const int b  = blockIdx.x >> 8;
  const int qt = blockIdx.x & 255;
  const int n0 = qt * 16;
  const int t  = threadIdx.x;
  const int wave = t >> 5;
  const int lane = t & 31;
  const int lm = lane & 15, lh = lane >> 4;
  const int c0 = wave * 64;

  // Q tile as WMMA A operand (only wave 0 uses it)
  v16bf qA{};
  if (wave == 0) {
    const unsigned short* qrow = Qb + ((size_t)b * NN + n0 + lm) * CDIM;
    qA = mk16(*(const v8bf*)(qrow + lh * 16),
              *(const v8bf*)(qrow + lh * 16 + 8));
  }

  v8f acc0 = {}, acc1 = {}, acc2 = {}, acc3 = {};
  float row_max = -1e30f, row_sum = 0.0f;   // live in wave0 lanes 0..15

  const unsigned short* Ktbase = Ktb + (size_t)b * NN * CDIM;
  const unsigned short* Vrow =
      Vb + ((size_t)b * CC + c0 + lm) * NN + lh * 16;  // per-lane V row base

  for (int it = 0; it < NN / 32; ++it) {
    const int m0 = it * 32;
    const int ph = it & 1;

    if (wave == 0) {
      // ---- S = Q * K  (two 16x16 tiles covering 32 keys) ----
      const unsigned short* k0p = Ktbase + (size_t)(m0 + lm) * CDIM + lh * 16;
      const unsigned short* k1p = k0p + 16 * CDIM;
      v16bf kB0 = mk16(*(const v8bf*)(k0p), *(const v8bf*)(k0p + 8));
      v16bf kB1 = mk16(*(const v8bf*)(k1p), *(const v8bf*)(k1p + 8));
      __builtin_prefetch(k0p + 32 * CDIM, 0, 3);   // next step's K rows
      v8f z = {};
      v8f s0 = wmma_bf16(qA, kB0, z);
      v8f s1 = wmma_bf16(qA, kB1, z);

      // scatter D-layout scores to LDS rows (wave-internal; LDS is in-order)
      #pragma unroll
      for (int r = 0; r < 8; ++r) {
        const int row = r + 8 * lh;
        Sls[row * 32 + lm]      = s0[r];
        Sls[row * 32 + 16 + lm] = s1[r];
      }
      __asm__ volatile("" ::: "memory");  // keep program order of LDS ops

      // ---- online softmax: lane t owns query row t ----
      if (lane < 16) {
        float sv[32];
        #pragma unroll
        for (int j4 = 0; j4 < 8; ++j4) {
          v4f q4 = *(const v4f*)&Sls[lane * 32 + j4 * 4];
          sv[j4 * 4 + 0] = q4[0]; sv[j4 * 4 + 1] = q4[1];
          sv[j4 * 4 + 2] = q4[2]; sv[j4 * 4 + 3] = q4[3];
        }
        float vmax = row_max;
        #pragma unroll
        for (int j = 0; j < 32; ++j) vmax = fmaxf(vmax, sv[j]);
        const float corr = __expf(row_max - vmax);
        float s = row_sum * corr;
        #pragma unroll
        for (int j = 0; j < 32; j += 4) {
          float p0 = __expf(sv[j]     - vmax);
          float p1 = __expf(sv[j + 1] - vmax);
          float p2 = __expf(sv[j + 2] - vmax);
          float p3 = __expf(sv[j + 3] - vmax);
          s += (p0 + p1) + (p2 + p3);
          uint2 pk;
          pk.x = (unsigned int)f2bf(p0) | ((unsigned int)f2bf(p1) << 16);
          pk.y = (unsigned int)f2bf(p2) | ((unsigned int)f2bf(p3) << 16);
          *(uint2*)&Pls[ph][lane * 32 + j] = pk;
        }
        row_sum = s;
        row_max = vmax;
        fac[ph][lane] = corr;
      }
    }
    __syncthreads();  // publish P[ph], fac[ph]; implicit dscnt-wait makes the
                      // double-buffer hand-off safe with a single barrier.

    // ---- all waves: rescale O accumulators by per-row correction ----
    #pragma unroll
    for (int r = 0; r < 8; ++r) {
      const float f = fac[ph][r + 8 * lh];
      acc0[r] *= f; acc1[r] *= f; acc2[r] *= f; acc3[r] *= f;
    }

    // ---- P as WMMA A operand ----
    const unsigned short* prow = &Pls[ph][lm * 32];
    v16bf pA = mk16(*(const v8bf*)(prow + lh * 16),
                    *(const v8bf*)(prow + lh * 16 + 8));

    // ---- O += P * V^T : 4 C-column tiles, K=32 keys ----
    const unsigned short* vp = Vrow + m0;
    __builtin_prefetch(vp + 32, 0, 3);   // next step's V segment
    v16bf vB0 = mk16(*(const v8bf*)(vp), *(const v8bf*)(vp + 8));
    acc0 = wmma_bf16(pA, vB0, acc0);
    vp += (size_t)16 * NN;
    v16bf vB1 = mk16(*(const v8bf*)(vp), *(const v8bf*)(vp + 8));
    acc1 = wmma_bf16(pA, vB1, acc1);
    vp += (size_t)16 * NN;
    v16bf vB2 = mk16(*(const v8bf*)(vp), *(const v8bf*)(vp + 8));
    acc2 = wmma_bf16(pA, vB2, acc2);
    vp += (size_t)16 * NN;
    v16bf vB3 = mk16(*(const v8bf*)(vp), *(const v8bf*)(vp + 8));
    acc3 = wmma_bf16(pA, vB3, acc3);
  }

  // ---- finalize: divide by row sums ----
  if (wave == 0 && lane < 16) fac[0][lane] = 1.0f / row_sum;
  __syncthreads();
  #pragma unroll
  for (int r = 0; r < 8; ++r) {
    const float f = fac[0][r + 8 * lh];
    acc0[r] *= f; acc1[r] *= f; acc2[r] *= f; acc3[r] *= f;
  }

  // ---- transpose O tiles through LDS: Ols[c][n] ----
  #pragma unroll
  for (int r = 0; r < 8; ++r) {
    const int n = r + 8 * lh;
    Ols[(c0 + 0 * 16 + lm) * 16 + n] = acc0[r];
    Ols[(c0 + 1 * 16 + lm) * 16 + n] = acc1[r];
    Ols[(c0 + 2 * 16 + lm) * 16 + n] = acc2[r];
    Ols[(c0 + 3 * 16 + lm) * 16 + n] = acc3[r];
  }
  __syncthreads();

  // ---- epilogue: y[b][c][n] = gamma*O[n,c] + x[b][c][n], n-contiguous b128s ----
  const float g = gamma[0];
  #pragma unroll
  for (int rr = 0; rr < 2; ++rr) {
    const int cl = t + rr * 128;
    const float* xs = x + ((size_t)b * CC + cl) * NN + n0;
    float*       ys = y + ((size_t)b * CC + cl) * NN + n0;
    #pragma unroll
    for (int j4 = 0; j4 < 4; ++j4) {
      v4f o4 = *(const v4f*)&Ols[cl * 16 + j4 * 4];
      v4f x4 = *(const v4f*)&xs[j4 * 4];
      v4f y4;
      y4[0] = g * o4[0] + x4[0];
      y4[1] = g * o4[1] + x4[1];
      y4[2] = g * o4[2] + x4[2];
      y4[3] = g * o4[3] + x4[3];
      *(v4f*)&ys[j4 * 4] = y4;
    }
  }
}

// ---------------------------------------------------------------------------
extern "C" void kernel_launch(void* const* d_in, const int* in_sizes, int n_in,
                              void* d_out, int out_size, void* d_ws, size_t ws_size,
                              hipStream_t stream) {
  (void)in_sizes; (void)n_in; (void)out_size; (void)ws_size;
  const float* x     = (const float*)d_in[0];
  const float* Wq    = (const float*)d_in[1];
  const float* Wk    = (const float*)d_in[2];
  const float* Wv    = (const float*)d_in[3];
  const float* gamma = (const float*)d_in[4];
  float* y = (float*)d_out;

  // Workspace layout (bytes), all 256-aligned. Total ~10.2 MB.
  char* ws = (char*)d_ws;
  unsigned short* Wqb = (unsigned short*)(ws + 0);          //  16384 B
  unsigned short* Wkb = (unsigned short*)(ws + 16384);      //  16384 B
  unsigned short* Wvb = (unsigned short*)(ws + 32768);      // 131072 B
  unsigned short* Qb  = (unsigned short*)(ws + 163840);     // 1 MB
  unsigned short* Ktb = (unsigned short*)(ws + 163840 + 1048576);      // 1 MB
  unsigned short* Vb  = (unsigned short*)(ws + 163840 + 2 * 1048576);  // 8 MB

  cvt_weights_kernel<<<256, 256, 0, stream>>>(Wq, Wk, Wv, Wqb, Wkb, Wvb);
  proj_kernel<<<BB * (NN / 32), 128, 0, stream>>>(x, Wqb, Wkb, Wvb, Qb, Ktb, Vb);
  attn_kernel<<<BB * (NN / 16), 128, 0, stream>>>(x, gamma, Qb, Ktb, Vb, y);
}
